// MemoryController_77068893160222
// MI455X (gfx1250) — matrix operand order repaired and checked by reading, not compiled, not run on hardware
//
#include <hip/hip_runtime.h>
#include <hip/hip_bf16.h>

#define NTHREADS 256

constexpr int TOK   = 512;   // L
constexpr int HID   = 64;    // H
constexpr int VOC   = 64;    // V
constexpr int HEADD = 32;    // HD
constexpr int KMEM  = 8;

// ---- LDS layout (bytes, dynamic shared) ----
constexpr int SMEM_A    = 0;        // 512*64 f16  = 65536   h / h1 / h2
constexpr int SMEM_B    = 65536;    // 131072 scratch: {q,k,vT} | ao2 | ffn-mid | f32 lm logits
constexpr int SMEM_C    = 196608;   // 512*64 f16  = 65536   attn-out / ffn2-out
constexpr int SMEM_ST   = 262144;   // 8 waves * 16*32 f16 = 8192  P staging
constexpr int SMEM_GSC  = 270336;   // 512 f32 = 2048 gate scores
constexpr int SMEM_QP   = 272384;   // 64 f32
constexpr int SMEM_RETR = 272640;   // 64 f32
constexpr int SMEM_LO   = 272896;   // 64 f32
constexpr int SMEM_RSC  = 273152;   // 8 f32
constexpr int SMEM_RW   = 273184;   // 8 f32
constexpr int SMEM_IDX  = 273216;   // 8 int
constexpr int SMEM_W    = 273248;   // 40960: f16 {W_in, W_out, lm_W} staged by TDM
constexpr int SMEM_TOTAL= 314208;

// ---- f16 weight pack offsets inside d_ws (element counts) ----
// LDS-staged weights first (contiguous TDM tile), then global-resident W1/W2.
constexpr int WOFF_IN  = 0;       // W_in  192x64  (12288)
constexpr int WOFF_OUT = 12288;   // W_out  64x64  ( 4096)
constexpr int WOFF_LM  = 16384;   // lm_W   64x64  ( 4096)
constexpr int WOFF_W1  = 20480;   // W1    128x64  ( 8192)
constexpr int WOFF_W2  = 28672;   // W2     64x128 ( 8192)
constexpr int WTOTAL   = 36864;
constexpr int WSTAGED_DWORDS = 10240;     // (12288+4096+4096)*2B / 4
constexpr int TASKBUF_OFF_BYTES = 73728;  // = WTOTAL*2, 256-aligned

typedef __attribute__((ext_vector_type(16))) _Float16 v16h;
typedef __attribute__((ext_vector_type(8)))  _Float16 v8h;
typedef __attribute__((ext_vector_type(4)))  _Float16 v4h;
typedef __attribute__((ext_vector_type(8)))  float    v8f;
typedef __attribute__((ext_vector_type(4)))  float    v4f;
typedef __attribute__((ext_vector_type(4)))  unsigned int v4u;
typedef __attribute__((ext_vector_type(8)))  int      v8i;
typedef __attribute__((ext_vector_type(4)))  int      v4i;

union Frag16 { v16h v; v8h h[2]; };

// A fragment (16x32 f16): lane r holds row r; k = i + hf*16
__device__ __forceinline__ v16h load_a_frag(const _Float16* rowBase, int hf) {
  Frag16 f;
  f.h[0] = *(const v8h*)(rowBase + hf * 16);
  f.h[1] = *(const v8h*)(rowBase + hf * 16 + 8);
  return f.v;
}
// B fragment (32x16 f16): lane r holds col n=r; k = i + hf*16 (16 contiguous halves)
__device__ __forceinline__ v16h load_b_frag(const _Float16* p16) {
  Frag16 f;
  f.h[0] = *(const v8h*)(p16);
  f.h[1] = *(const v8h*)(p16 + 8);
  return f.v;
}
__device__ __forceinline__ v8f wmma16(v16h a, v16h b, v8f c) {
  return __builtin_amdgcn_wmma_f32_16x16x32_f16(false, a, false, b, (short)0, c,
                                                false, false);
}

#if __has_builtin(__builtin_amdgcn_tensor_load_to_lds) && \
    __has_builtin(__builtin_amdgcn_s_wait_tensorcnt)
#define HAVE_TDM 1
#endif

#ifdef HAVE_TDM
// TDM: contiguous 1-D copy of nDwords dwords from global gsrc -> LDS ldsByteOff.
// D# packed per CDNA5 ISA §8.3/8.4 (count=1, type=2, data_size=4B,
// tensor_dim0 = tile_dim0 = nDwords, tensor_dim1 = tile_dim1 = 1).
__device__ __forceinline__ void tdm_load_pack(const _Float16* gsrc,
                                              unsigned ldsByteOff,
                                              unsigned nDwords) {
  const unsigned long long ga = (unsigned long long)gsrc;
  v4u g0;
  g0[0] = 1u;                                            // count=1, user mode
  g0[1] = ldsByteOff;                                    // lds_addr
  g0[2] = (unsigned)ga;                                  // global_addr[31:0]
  g0[3] = (unsigned)((ga >> 32) & 0x01FFFFFFu) | (2u << 30);  // addr[56:32]|type=2
  v8i g1;
  g1[0] = (int)(2u << 16);                               // data_size: 4 bytes
  g1[1] = (int)((nDwords & 0xFFFFu) << 16);              // tensor_dim0 [63:48]
  g1[2] = (int)(((nDwords >> 16) & 0xFFFFu) | (1u << 16)); // dim0 hi | tensor_dim1=1
  g1[3] = (int)((nDwords & 0xFFFFu) << 16);              // tile_dim0 [127:112]
  g1[4] = 1;                                             // tile_dim1=1, tile_dim2=0
  g1[5] = (int)nDwords;                                  // tensor_dim0_stride lo
  g1[6] = 0;
  g1[7] = 0;
  v4i g2 = {0, 0, 0, 0};
  v4i g3 = {0, 0, 0, 0};
#if __clang_major__ >= 23
  v8i g4 = {0, 0, 0, 0, 0, 0, 0, 0};
  __builtin_amdgcn_tensor_load_to_lds(g0, g1, g2, g3, g4, 0);
#else
  __builtin_amdgcn_tensor_load_to_lds(g0, g1, g2, g3, 0);
#endif
}
#endif

// Tiled GEMM: C[512 x (16*ntiles)] = A[512 x (32*ksteps)] * W^T (+bias, opt relu)
// A in LDS row-major (stride lda halves); W rows (output n) at stride ldw
// (either LDS-staged or global f16).
__device__ __forceinline__ void gemm_tiles(
    const _Float16* __restrict__ A, int lda,
    const _Float16* __restrict__ W, int ldw,
    const float* __restrict__ bias,
    _Float16* out16, float* out32, int ldo,
    int ntiles, int ksteps, bool doRelu, int wave, int lane) {
  const int hf = lane >> 4, rr = lane & 15;
  const int njobs = 32 * ntiles;
  for (int job = wave; job < njobs; job += 8) {
    const int mt = job / ntiles;
    const int nt = job % ntiles;
    v8f c = {};
    const _Float16* arow = A + (mt * 16 + rr) * lda;
    const _Float16* wrow = W + (nt * 16 + rr) * ldw + hf * 16;
    for (int ks = 0; ks < ksteps; ks++)
      c = wmma16(load_a_frag(arow + ks * 32, hf), load_b_frag(wrow + ks * 32), c);
    const float bv = bias ? bias[nt * 16 + rr] : 0.0f;
    for (int j = 0; j < 8; j++) {
      float v = c[j] + bv;
      if (doRelu) v = fmaxf(v, 0.0f);
      const int m = mt * 16 + j + hf * 8;
      const int n = nt * 16 + rr;
      if (out16) out16[m * ldo + n] = (_Float16)v;
      else       out32[m * ldo + n] = v;
    }
  }
}

// LayerNorm over rows: io <- LN(io + add) * g + b   (f16 in/out, f32 math)
__device__ __forceinline__ void layernorm_rows(
    _Float16* io, const _Float16* add,
    const float* __restrict__ g, const float* __restrict__ bta, int tid) {
  for (int t = tid; t < TOK; t += NTHREADS) {
    const _Float16* a = io + t * HID;
    const _Float16* c = add + t * HID;
    float s = 0.f, ss = 0.f;
    for (int i = 0; i < HID; i++) {
      float v = (float)a[i] + (float)c[i];
      s += v; ss += v * v;
    }
    const float mean = s * (1.0f / HID);
    const float var  = ss * (1.0f / HID) - mean * mean;
    const float rstd = rsqrtf(var + 1e-5f);
    _Float16* w = io + t * HID;
    for (int i = 0; i < HID; i++) {
      float v = (float)a[i] + (float)c[i];
      w[i] = (_Float16)((v - mean) * rstd * g[i] + bta[i]);
    }
  }
}

// -------- weight f32 -> f16 pack --------
__global__ void prep_weights(const float* __restrict__ Win,
                             const float* __restrict__ Wout,
                             const float* __restrict__ W1,
                             const float* __restrict__ W2,
                             const float* __restrict__ lmW,
                             _Float16* __restrict__ dst) {
  const int i = blockIdx.x * blockDim.x + threadIdx.x;
  if (i < WOFF_OUT)      dst[i] = (_Float16)Win[i - WOFF_IN];
  else if (i < WOFF_LM)  dst[i] = (_Float16)Wout[i - WOFF_OUT];
  else if (i < WOFF_W1)  dst[i] = (_Float16)lmW[i - WOFF_LM];
  else if (i < WOFF_W2)  dst[i] = (_Float16)W1[i - WOFF_W1];
  else if (i < WTOTAL)   dst[i] = (_Float16)W2[i - WOFF_W2];
}

// -------- fused per-batch-element pipeline: one workgroup per b --------
__global__ __launch_bounds__(NTHREADS) void fused_memctrl(
    const float* __restrict__ embed,  const float* __restrict__ qembed,
    const float* __restrict__ b_in,   const float* __restrict__ b_out,
    const float* __restrict__ ln1_g,  const float* __restrict__ ln1_b,
    const float* __restrict__ b1,     const float* __restrict__ b2,
    const float* __restrict__ ln2_g,  const float* __restrict__ ln2_b,
    const float* __restrict__ lm_b,   const float* __restrict__ gate_W,
    const float* __restrict__ gate_b, const float* __restrict__ qp_W,
    const float* __restrict__ qp_b,   const float* __restrict__ op_W,
    const float* __restrict__ op_b,
    const int* __restrict__ seq, const int* __restrict__ query,
    const int* __restrict__ target,
    const _Float16* __restrict__ w16,
    float* __restrict__ taskBuf, float* __restrict__ out) {
  extern __shared__ char smem[];
  const int b    = blockIdx.x;
  const int tid  = threadIdx.x;
  // wave index is uniform across the wave: pin it to an SGPR so job loops and
  // q/k/v selection lower to scalar branches instead of EXEC-masked ladders.
  const int wave = __builtin_amdgcn_readfirstlane(tid >> 5);
  const int lane = tid & 31;
  const int hf   = lane >> 4;
  const int rr   = lane & 15;

  _Float16* sA     = (_Float16*)(smem + SMEM_A);
  _Float16* sBh    = (_Float16*)(smem + SMEM_B);
  _Float16* sC     = (_Float16*)(smem + SMEM_C);
  _Float16* stage  = (_Float16*)(smem + SMEM_ST);
  float*    gsc    = (float*)(smem + SMEM_GSC);
  float*    qpS    = (float*)(smem + SMEM_QP);
  float*    retr   = (float*)(smem + SMEM_RETR);
  float*    loS    = (float*)(smem + SMEM_LO);
  float*    rsc    = (float*)(smem + SMEM_RSC);
  float*    rw     = (float*)(smem + SMEM_RW);
  int*      selIdx = (int*)(smem + SMEM_IDX);
  _Float16* sW     = (_Float16*)(smem + SMEM_W);

  // ---- Phase 0a: kick off TDM weight stage (W_in|W_out|lm_W -> LDS) ----
#ifdef HAVE_TDM
  if (wave == 0) tdm_load_pack(w16, SMEM_W, WSTAGED_DWORDS);
#else
  {
    const unsigned* src = (const unsigned*)w16;
    unsigned* dstw = (unsigned*)sW;
    for (int i = tid; i < WSTAGED_DWORDS; i += NTHREADS) dstw[i] = src[i];
  }
#endif
  // warm L2 for the global-resident W1/W2 fragments
  for (int off = tid * 256; off < 32768; off += NTHREADS * 256)
    __builtin_prefetch((const char*)(w16 + WOFF_W1) + off, 0, 0);

  // ---- Phase 0b: h = embed[seq[b]] -> sA (f16), overlapped with the TDM ----
  for (int t = tid; t < TOK; t += NTHREADS) {
    const v4f* e = (const v4f*)(embed + seq[b * TOK + t] * HID);
    v4h* d = (v4h*)(sA + t * HID);
    for (int i = 0; i < 16; i++) {
      const v4f x = e[i];
      v4h y;
      y[0] = (_Float16)x[0]; y[1] = (_Float16)x[1];
      y[2] = (_Float16)x[2]; y[3] = (_Float16)x[3];
      d[i] = y;
    }
  }
#ifdef HAVE_TDM
  if (wave == 0) __builtin_amdgcn_s_wait_tensorcnt(0);
#endif
  __syncthreads();

  // ---- Phase 1: per-head QKV projection + flash attention -> sC ----
  _Float16* qv = sBh;                     // 512 x 32
  _Float16* kv = sBh + TOK * HEADD;       // 512 x 32
  _Float16* vT = sBh + 2 * TOK * HEADD;   // 32 x 512 (transposed V)
  for (int head = 0; head < 2; head++) {
    // q/k/v slices of the 3H QKV GEMM for this head (6 N-tiles x 32 M-tiles)
    for (int job = wave; job < 32 * 6; job += 8) {
      const int mt = job / 6, rem = job % 6;
      const int which = rem >> 1, nsub = rem & 1;     // 0=q 1=k 2=v
      const int nbase = which * 64 + head * HEADD + nsub * 16;
      v8f c = {};
      const _Float16* arow = sA + (mt * 16 + rr) * HID;
      const _Float16* wrow = sW + WOFF_IN + (nbase + rr) * HID + hf * 16;
      for (int ks = 0; ks < 2; ks++)
        c = wmma16(load_a_frag(arow + ks * 32, hf), load_b_frag(wrow + ks * 32), c);
      const float bv = b_in[nbase + rr];
      // uniform (scalar) selection of destination + strides
      _Float16* dst; int mm, nn;
      if (which == 0)      { dst = qv; mm = HEADD; nn = 1; }
      else if (which == 1) { dst = kv; mm = HEADD; nn = 1; }
      else                 { dst = vT; mm = 1;     nn = TOK; }
      const int ncol = nsub * 16 + rr;
      for (int j = 0; j < 8; j++) {
        const int m = mt * 16 + j + hf * 8;
        dst[m * mm + ncol * nn] = (_Float16)(c[j] + bv);
      }
    }
    __syncthreads();

    // flash attention: one wave per 16-query tile
    const float invsq = 0.17677669529663687f;   // 1/sqrt(HD=32)
    for (int qt = wave; qt < 32; qt += 8) {
      v16h aq = load_a_frag(qv + (qt * 16 + rr) * HEADD, hf);
      float mrow[8], lrow[8];
      v8f o0 = {}, o1 = {};
      for (int j = 0; j < 8; j++) { mrow[j] = -1e30f; lrow[j] = 0.0f; }
      _Float16* stg = stage + wave * (16 * 32);
      for (int kp = 0; kp < 16; kp++) {          // 32 keys per iteration
        v16h bk0 = load_b_frag(kv + (kp * 32 + rr) * HEADD + hf * 16);
        v16h bk1 = load_b_frag(kv + (kp * 32 + 16 + rr) * HEADD + hf * 16);
        v8f z = {};
        v8f s0 = wmma16(aq, bk0, z);
        v8f s1 = wmma16(aq, bk1, z);
        for (int j = 0; j < 8; j++) {
          float v0 = s0[j] * invsq, v1 = s1[j] * invsq;
          float mx = fmaxf(v0, v1);
          for (int m = 1; m < 16; m <<= 1) mx = fmaxf(mx, __shfl_xor(mx, m));
          const float mnew  = fmaxf(mrow[j], mx);
          const float alpha = __expf(mrow[j] - mnew);
          const float p0 = __expf(v0 - mnew);
          const float p1 = __expf(v1 - mnew);
          float ps = p0 + p1;
          for (int m = 1; m < 16; m <<= 1) ps += __shfl_xor(ps, m);
          lrow[j] = lrow[j] * alpha + ps;
          mrow[j] = mnew;
          o0[j] *= alpha; o1[j] *= alpha;
          stg[(j + hf * 8) * 32 + rr]      = (_Float16)p0;   // same-wave LDS:
          stg[(j + hf * 8) * 32 + 16 + rr] = (_Float16)p1;   // DS ops in-order
        }
        v16h ap  = load_a_frag(stg + rr * 32, hf);
        v16h bv0 = load_b_frag(vT + rr * TOK + kp * 32 + hf * 16);
        v16h bv1 = load_b_frag(vT + (16 + rr) * TOK + kp * 32 + hf * 16);
        o0 = wmma16(ap, bv0, o0);
        o1 = wmma16(ap, bv1, o1);
      }
      for (int j = 0; j < 8; j++) {
        const int row = qt * 16 + j + hf * 8;
        const float inv = 1.0f / lrow[j];
        sC[row * HID + head * HEADD + rr]      = (_Float16)(o0[j] * inv);
        sC[row * HID + head * HEADD + 16 + rr] = (_Float16)(o1[j] * inv);
      }
    }
    __syncthreads();
  }

  // ---- Phase 2: W_out projection (sC -> sB[0:64K)) ----
  gemm_tiles(sC, HID, sW + WOFF_OUT, HID, b_out, sBh, nullptr, HID, 4, 2,
             false, wave, lane);
  __syncthreads();
  // ---- LN1: sA <- LN(h + ao) ----
  layernorm_rows(sA, sBh, ln1_g, ln1_b, tid);
  __syncthreads();

  // ---- Phase 3: FFN ----
  gemm_tiles(sA, HID, w16 + WOFF_W1, HID, b1, sBh, nullptr, 128, 8, 2,
             true, wave, lane);                       // relu(h1 @ W1^T) -> 512x128
  __syncthreads();
  gemm_tiles(sBh, 128, w16 + WOFF_W2, 128, b2, sC, nullptr, HID, 4, 4,
             false, wave, lane);                      // -> 512x64 in sC
  __syncthreads();
  layernorm_rows(sA, sC, ln2_g, ln2_b, tid);          // sA = h2
  __syncthreads();

  // ---- Phase 4: LM logits (f32, 512x64) into sB ----
  gemm_tiles(sA, HID, sW + WOFF_LM, HID, lm_b, nullptr, (float*)sBh, VOC, 4, 2,
             false, wave, lane);
  __syncthreads();

  // ---- Phase 5: per-token LM loss + gate scores ----
  const float* logitsF = (const float*)sBh;
  for (int t = tid; t < TOK; t += NTHREADS) {
    const _Float16* hrow = sA + t * HID;
    float g = gate_b[0];
    for (int i = 0; i < HID; i++) g += (float)hrow[i] * gate_W[i];
    gsc[t] = g;
    float o = 0.0f;
    if (t != TOK - 1) {
      const float* lr = logitsF + t * VOC;
      float mx = lr[0];
      for (int v = 1; v < VOC; v++) mx = fmaxf(mx, lr[v]);
      float s = 0.f;
      for (int v = 0; v < VOC; v++) s += __expf(lr[v] - mx);
      o = (mx + __logf(s)) - lr[seq[b * TOK + t + 1]];
    }
    out[1 + b * TOK + t] = o;
  }
  __syncthreads();

  // ---- Phase 6: top-8 gate selection (wave 0) ----
  if (wave == 0) {
    for (int sel = 0; sel < KMEM; sel++) {
      float best = -1e30f; int bi = 0;
      for (int t = lane; t < TOK; t += 32) {
        const float v = gsc[t];
        if (v > best) { best = v; bi = t; }
      }
      for (int m = 16; m >= 1; m >>= 1) {
        const float ov = __shfl_xor(best, m);
        const int   oi = __shfl_xor(bi, m);
        if (ov > best || (ov == best && oi < bi)) { best = ov; bi = oi; }
      }
      if (lane == 0) { selIdx[sel] = bi; gsc[bi] = -1e30f; }
    }
  }
  __syncthreads();

  // ---- Phase 7: memory reader ----
  if (tid < HID) {                       // qp = qembed[query] @ qp_W^T + qp_b
    const float* qe = qembed + query[b] * HID;
    const float* wr = qp_W + tid * HID;
    float acc = qp_b[tid];
    for (int i = 0; i < HID; i++) acc += qe[i] * wr[i];
    qpS[tid] = acc;
  }
  __syncthreads();
  if (tid < KMEM) {                      // rscores
    const _Float16* mk = sA + selIdx[tid] * HID;
    float acc = 0.f;
    for (int i = 0; i < HID; i++) acc += qpS[i] * (float)mk[i];
    rsc[tid] = acc * 0.125f;             // 1/sqrt(H=64)
  }
  __syncthreads();
  if (tid == 0) {                        // softmax over K
    float mx = rsc[0];
    for (int k = 1; k < KMEM; k++) mx = fmaxf(mx, rsc[k]);
    float s = 0.f;
    for (int k = 0; k < KMEM; k++) { rw[k] = __expf(rsc[k] - mx); s += rw[k]; }
    const float inv = 1.0f / s;
    for (int k = 0; k < KMEM; k++) rw[k] *= inv;
  }
  __syncthreads();
  if (tid < HID) {                       // retrieved
    float acc = 0.f;
    for (int k = 0; k < KMEM; k++) acc += rw[k] * (float)sA[selIdx[k] * HID + tid];
    retr[tid] = acc;
  }
  __syncthreads();
  if (tid < VOC) {                       // output logits
    const float* wr = op_W + tid * HID;
    float acc = op_b[tid];
    for (int i = 0; i < HID; i++) acc += retr[i] * wr[i];
    loS[tid] = acc;
  }
  __syncthreads();
  if (tid == 0) {                        // per-b task loss
    float mx = loS[0];
    for (int v = 1; v < VOC; v++) mx = fmaxf(mx, loS[v]);
    float s = 0.f;
    for (int v = 0; v < VOC; v++) s += __expf(loS[v] - mx);
    taskBuf[b] = (mx + __logf(s)) - loS[target[b]];
  }
}

// -------- deterministic mean over 256 per-b losses -> d_out[0] --------
__global__ void reduce_task(const float* __restrict__ taskBuf,
                            float* __restrict__ out) {
  __shared__ float s[NTHREADS];
  const int tid = threadIdx.x;
  s[tid] = taskBuf[tid];
  __syncthreads();
  for (int st = NTHREADS / 2; st > 0; st >>= 1) {
    if (tid < st) s[tid] += s[tid + st];
    __syncthreads();
  }
  if (tid == 0) out[0] = s[0] * (1.0f / 256.0f);
}

extern "C" void kernel_launch(void* const* d_in, const int* in_sizes, int n_in,
                              void* d_out, int out_size, void* d_ws, size_t ws_size,
                              hipStream_t stream) {
  (void)in_sizes; (void)n_in; (void)out_size; (void)ws_size;
  const float* embed  = (const float*)d_in[0];
  const float* qembed = (const float*)d_in[1];
  const float* W_in   = (const float*)d_in[2];
  const float* b_in   = (const float*)d_in[3];
  const float* W_out  = (const float*)d_in[4];
  const float* b_out  = (const float*)d_in[5];
  const float* ln1_g  = (const float*)d_in[6];
  const float* ln1_b  = (const float*)d_in[7];
  const float* W1     = (const float*)d_in[8];
  const float* b1     = (const float*)d_in[9];
  const float* W2     = (const float*)d_in[10];
  const float* b2     = (const float*)d_in[11];
  const float* ln2_g  = (const float*)d_in[12];
  const float* ln2_b  = (const float*)d_in[13];
  const float* lm_W   = (const float*)d_in[14];
  const float* lm_b   = (const float*)d_in[15];
  const float* gate_W = (const float*)d_in[16];
  const float* gate_b = (const float*)d_in[17];
  const float* qp_W   = (const float*)d_in[18];
  const float* qp_b   = (const float*)d_in[19];
  const float* op_W   = (const float*)d_in[20];
  const float* op_b   = (const float*)d_in[21];
  const int*   seq    = (const int*)d_in[22];
  const int*   query  = (const int*)d_in[23];
  const int*   target = (const int*)d_in[24];

  _Float16* w16     = (_Float16*)d_ws;
  float*    taskBuf = (float*)((char*)d_ws + TASKBUF_OFF_BYTES);

  prep_weights<<<(WTOTAL + NTHREADS - 1) / NTHREADS, NTHREADS, 0, stream>>>(
      W_in, W_out, W1, W2, lm_W, w16);
  fused_memctrl<<<256, NTHREADS, SMEM_TOTAL, stream>>>(
      embed, qembed, b_in, b_out, ln1_g, ln1_b, b1, b2, ln2_g, ln2_b, lm_b,
      gate_W, gate_b, qp_W, qp_b, op_W, op_b, seq, query, target, w16,
      taskBuf, (float*)d_out);
  reduce_task<<<1, NTHREADS, 0, stream>>>(taskBuf, (float*)d_out);
}
